// TSMamba_41111426957712
// MI455X (gfx1250) — matrix operand order repaired and checked by reading, not compiled, not run on hardware
//
#include <hip/hip_runtime.h>
#include <hip/hip_bf16.h>
#include <math.h>

// ---------------------------------------------------------------------------
// TSMamba block for MI455X (gfx1250).
//  - All dense projections AND both 3x3x3 convs (implicit GEMM, 27 taps)
//    run on v_wmma_f32_16x16x32_bf16 with fp32 accumulation.
//  - LDS tiles hold bf16 PAIRS in 32-bit words, k-contiguous for both
//    operands, so fragments load as 8x ds_load_b32 each (matches the
//    ISA 7.12.2 VGPR pair layout exactly).
//  - Tile staging branches uniformly on a full-tile predicate: the hot
//    path has no per-element exec masking.
// ---------------------------------------------------------------------------

typedef __attribute__((ext_vector_type(16))) __bf16 v16bf;
typedef __attribute__((ext_vector_type(8)))  float  v8f;

#define CDIM    96
#define LTOK    4096          // 16^3
#define BSZ     2
#define DINNER  192
#define DSTATE  16
#define DTRANK  6

#define EP_NONE          0
#define EP_BIAS          1
#define EP_BIAS_GELU     2
#define EP_BIAS_SOFTPLUS 3
#define EP_RES           4
#define EP_BIAS_RES      5

__device__ __forceinline__ unsigned int f2bf(float f) {
  unsigned int u = __float_as_uint(f);
  u += 0x7FFFu + ((u >> 16) & 1u);          // round-to-nearest-even
  return u >> 16;
}
__device__ __forceinline__ unsigned int packbf(float lo, float hi) {
  return f2bf(lo) | (f2bf(hi) << 16);
}

union FragU { unsigned int u[8]; v16bf v; };

// fragment gather from LDS (k-contiguous pair words)
__device__ __forceinline__ v8f wmma_step(const unsigned int As[32][17],
                                         const unsigned int Bs[32][17],
                                         int arow, int brow, int half, v8f acc)
{
  FragU af, bf;
#pragma unroll
  for (int j = 0; j < 8; ++j) {
    // A 16x32: lanes 0-15 K{0..7,16..23}, lanes 16-31 K{8..15,24..31}; pairs
    af.u[j] = As[arow][((j < 4) ? 0 : 8) + half * 4 + (j & 3)];
    // B 32x16: lanes 0-15 K{0..7,16..23}, lanes 16-31 K{8..15,24..31}; pairs
    bf.u[j] = Bs[brow][((j < 4) ? 0 : 8) + half * 4 + (j & 3)];
  }
  return __builtin_amdgcn_wmma_f32_16x16x32_bf16(false, af.v, false, bf.v,
                                                 (short)0, acc, false, false);
}

// ---------------------------------------------------------------------------
// Generic strided-batched GEMM: C[b] = A[b](MxK) * B[b](KxN)  (+ epilogue)
// 128 threads = 4 waves (2x2), 32x32 macro-tile, K-step 32.
// ---------------------------------------------------------------------------
__global__ __launch_bounds__(128)
void k_gemm_wmma(const float* __restrict__ A, long sAm, long sAk, long sAb,
                 const float* __restrict__ B, long sBk, long sBn, long sBb,
                 float* __restrict__ C, long sCm, long sCn, long sCb,
                 const float* __restrict__ bias,
                 const float* __restrict__ res, long sRb,
                 int M, int N, int K, int ep)
{
  __shared__ unsigned int As[32][17];     // [m][k-pair]
  __shared__ unsigned int Bs[32][17];     // [n][k-pair]  (transposed stage)

  const int b = blockIdx.z;
  A += (long)b * sAb;
  B += (long)b * sBb;
  C += (long)b * sCb;
  if (res) res += (long)b * sRb;

  const int tid  = threadIdx.x;
  const int wave = tid >> 5, lane = tid & 31;
  const int wm   = wave >> 1, wn = wave & 1;
  const int m0   = blockIdx.y * 32, n0 = blockIdx.x * 32;
  const int half = lane >> 4, l15 = lane & 15;
  const bool mnfull = (m0 + 32 <= M) && (n0 + 32 <= N);

  v8f acc = {};

  for (int k0 = 0; k0 < K; k0 += 32) {
    if (mnfull && (k0 + 32 <= K)) {
      // ---- fast path: no bounds, straight-line loads ----
      for (int i = tid; i < 512; i += 128) {
        int rA = i >> 4, cA = i & 15;                 // A: row-major tasks
        long ab = (long)(m0 + rA) * sAm + (long)(k0 + 2 * cA) * sAk;
        As[rA][cA] = packbf(A[ab], A[ab + sAk]);
        int nB = i & 31, cB = i >> 5;                 // B: n fastest => coalesced
        long bb = (long)(k0 + 2 * cB) * sBk + (long)(n0 + nB) * sBn;
        Bs[nB][cB] = packbf(B[bb], B[bb + sBk]);
      }
    } else {
      // ---- edge path (rare): per-element zero-fill ----
      for (int i = tid; i < 512; i += 128) {
        int rA = i >> 4, cA = i & 15;
        int m = m0 + rA, k = k0 + 2 * cA;
        float a0 = (m < M && k     < K) ? A[(long)m * sAm + (long)k * sAk] : 0.f;
        float a1 = (m < M && k + 1 < K) ? A[(long)m * sAm + (long)(k + 1) * sAk] : 0.f;
        As[rA][cA] = packbf(a0, a1);
        int nB = i & 31, cB = i >> 5;
        int n = n0 + nB, kk = k0 + 2 * cB;
        float b0 = (n < N && kk     < K) ? B[(long)kk * sBk + (long)n * sBn] : 0.f;
        float b1 = (n < N && kk + 1 < K) ? B[(long)(kk + 1) * sBk + (long)n * sBn] : 0.f;
        Bs[nB][cB] = packbf(b0, b1);
      }
    }
    __syncthreads();
    acc = wmma_step(As, Bs, wm * 16 + l15, wn * 16 + l15, half, acc);
    __syncthreads();
  }

  const int cn = n0 + wn * 16 + l15;
  if (cn < N) {
#pragma unroll
    for (int r = 0; r < 8; ++r) {
      int cm = m0 + wm * 16 + r + 8 * half;
      if (cm < M) {
        float v = acc[r];
        if (ep == EP_BIAS || ep == EP_BIAS_GELU ||
            ep == EP_BIAS_SOFTPLUS || ep == EP_BIAS_RES)
          v += bias[cm];
        if (ep == EP_BIAS_GELU)
          v = 0.5f * v * (1.0f + erff(v * 0.70710678118654752f));
        if (ep == EP_BIAS_SOFTPLUS)
          v = (v > 20.0f) ? v : log1pf(__expf(v));
        if (ep == EP_RES || ep == EP_BIAS_RES)
          v += res[(long)cm * sCm + (long)cn * sCn];
        C[(long)cm * sCm + (long)cn * sCn] = v;
      }
    }
  }
}

// ---------------------------------------------------------------------------
// 3x3x3 conv (pad=1) as implicit GEMM on WMMA: 27 taps x ceil(Cin/32)
// k-steps.  B operand = spatially shifted input columns with per-element
// boundary zeroing done at bf16-staging time.  X: (B,Cin,16,16,16),
// W: (Cout,Cin,3,3,3), out: (B,Cout,L).
// ---------------------------------------------------------------------------
__global__ __launch_bounds__(128)
void k_conv3_wmma(const float* __restrict__ X, const float* __restrict__ W,
                  const float* __restrict__ bias, float* __restrict__ out,
                  int Cin, int Cout)
{
  __shared__ unsigned int As[32][17];     // [co][ci-pair]
  __shared__ unsigned int Bs[32][17];     // [n][ci-pair]

  const int b  = blockIdx.z;
  const int n0 = blockIdx.x * 32;         // spatial tile (32 voxels)
  const int m0 = blockIdx.y * 32;         // cout tile
  const int tid  = threadIdx.x;
  const int wave = tid >> 5, lane = tid & 31;
  const int wm   = wave >> 1, wn = wave & 1;
  const int half = lane >> 4, l15 = lane & 15;
  const int kSteps = (Cin + 31) >> 5;

  v8f acc = {};

  for (int tap = 0; tap < 27; ++tap) {
    const int dz = tap / 9 - 1, dy = (tap / 3) % 3 - 1, dx = tap % 3 - 1;
    const int off = dz * 256 + dy * 16 + dx;
    for (int ks = 0; ks < kSteps; ++ks) {
      const int k0 = ks << 5;
      for (int i = tid; i < 512; i += 128) {
        // A tile: W[m][k][tap], stride 27 over (m*Cin+k)
        int rA = i >> 4, cA = i & 15;
        int m = m0 + rA, k = k0 + 2 * cA;
        float a0 = (m < Cout && k     < Cin) ? W[((long)m * Cin + k) * 27 + tap] : 0.f;
        float a1 = (m < Cout && k + 1 < Cin) ? W[((long)m * Cin + k + 1) * 27 + tap] : 0.f;
        As[rA][cA] = packbf(a0, a1);
        // B tile: X[b][k][s+off] with boundary zeroing
        int nB = i & 31, cB = i >> 5;
        int s = n0 + nB, kk = k0 + 2 * cB;
        int xx = (s & 15) + dx, yy = ((s >> 4) & 15) + dy, zz = (s >> 8) + dz;
        bool valid = ((unsigned)xx < 16u) & ((unsigned)yy < 16u) & ((unsigned)zz < 16u);
        float b0 = 0.f, b1 = 0.f;
        if (valid) {
          long base = (long)b * Cin * LTOK + (s + off);
          if (kk     < Cin) b0 = X[base + (long)kk * LTOK];
          if (kk + 1 < Cin) b1 = X[base + (long)(kk + 1) * LTOK];
        }
        Bs[nB][cB] = packbf(b0, b1);
      }
      __syncthreads();
      acc = wmma_step(As, Bs, wm * 16 + l15, wn * 16 + l15, half, acc);
      __syncthreads();
    }
  }

  const int cn = n0 + wn * 16 + l15;
#pragma unroll
  for (int r = 0; r < 8; ++r) {
    int cm = m0 + wm * 16 + r + 8 * half;
    if (cm < Cout)
      out[((long)b * Cout + cm) * LTOK + cn] = acc[r] + bias[cm];
  }
}

// ---------------------------------------------------------------------------
// GroupNorm over (B, C, L): one block per (b, group); optional ReLU and
// post-activation residual add.  In-place safe.
// ---------------------------------------------------------------------------
__global__ __launch_bounds__(256)
void k_groupnorm(const float* __restrict__ in, float* __restrict__ out,
                 const float* __restrict__ w, const float* __restrict__ bias,
                 const float* __restrict__ res, int C, int groups, int relu)
{
  int b = blockIdx.x / groups, g = blockIdx.x % groups;
  int cpg = C / groups;
  long base = ((long)b * C + (long)g * cpg) * LTOK;
  int n = cpg * LTOK;
  float s = 0.f, s2 = 0.f;
  for (int i = threadIdx.x; i < n; i += 256) {
    float v = in[base + i]; s += v; s2 += v * v;
  }
  __shared__ float red0[256], red1[256];
  red0[threadIdx.x] = s; red1[threadIdx.x] = s2;
  __syncthreads();
  for (int off = 128; off > 0; off >>= 1) {
    if (threadIdx.x < off) {
      red0[threadIdx.x] += red0[threadIdx.x + off];
      red1[threadIdx.x] += red1[threadIdx.x + off];
    }
    __syncthreads();
  }
  float mu   = red0[0] / (float)n;
  float var  = red1[0] / (float)n - mu * mu;
  float rstd = rsqrtf(var + 1e-5f);
  for (int i = threadIdx.x; i < n; i += 256) {
    int c = g * cpg + i / LTOK;
    float v = (in[base + i] - mu) * rstd * w[c] + bias[c];
    if (relu) v = fmaxf(v, 0.f);
    if (res)  v += res[base + i];
    out[base + i] = v;
  }
}

// LayerNorm over C per token: in (B,C,L) -> out (B,L,C)
__global__ __launch_bounds__(256)
void k_layernorm(const float* __restrict__ in, const float* __restrict__ w,
                 const float* __restrict__ bias, float* __restrict__ out)
{
  int idx = blockIdx.x * blockDim.x + threadIdx.x;
  if (idx >= BSZ * LTOK) return;
  int b = idx / LTOK, l = idx % LTOK;
  const float* ip = in + (long)b * CDIM * LTOK + l;
  float s = 0.f, s2 = 0.f;
  for (int c = 0; c < CDIM; ++c) { float v = ip[(long)c * LTOK]; s += v; s2 += v * v; }
  float mu = s / CDIM, var = s2 / CDIM - mu * mu, rstd = rsqrtf(var + 1e-5f);
  float* op = out + (long)idx * CDIM;
  for (int c = 0; c < CDIM; ++c)
    op[c] = (ip[(long)c * LTOK] - mu) * rstd * w[c] + bias[c];
}

// direction index map: sequence position t -> storage column
__device__ __forceinline__ int src_of(int t, int dir) {
  if (dir == 0) return t;                       // fwd
  if (dir == 1) return LTOK - 1 - t;            // bwd
  return (t & 3) * 1024 + (t >> 2);             // slice-interleave (NSLICES=4)
}

// Depthwise causal conv1d (K=4, left pad 3) + SiLU, per direction.
__global__ __launch_bounds__(256)
void k_dwconv_silu(const float* __restrict__ xz, const float* __restrict__ cw,
                   const float* __restrict__ cb, float* __restrict__ xc, int dir)
{
  int idx = blockIdx.x * blockDim.x + threadIdx.x;
  if (idx >= BSZ * DINNER * LTOK) return;
  int t = idx % LTOK;
  int d = (idx / LTOK) % DINNER;
  int b = idx / (LTOK * DINNER);
  const float* row = xz + ((long)b * 2 * DINNER + d) * LTOK;
  float acc = cb[d];
#pragma unroll
  for (int j = 0; j < 4; ++j) {
    int tt = t - 3 + j;
    if (tt >= 0) acc = fmaf(row[src_of(tt, dir)], cw[d * 4 + j], acc);
  }
  acc = acc / (1.0f + __expf(-acc));            // SiLU
  xc[((long)b * DINNER + d) * LTOK + src_of(t, dir)] = acc;
}

// Selective scan: one lane per (b,d,n); 16-lane shuffle reduction per step.
__global__ __launch_bounds__(256)
void k_scan(const float* __restrict__ xc, const float* __restrict__ xdbl,
            const float* __restrict__ dtf, const float* __restrict__ xz,
            const float* __restrict__ A_log, const float* __restrict__ Dp,
            float* __restrict__ ysum, int dir)
{
  int g = blockIdx.x * blockDim.x + threadIdx.x;
  if (g >= BSZ * DINNER * DSTATE) return;       // exact fit: never taken
  int n = g & 15;
  int d = (g >> 4) % DINNER;
  int b = g / (16 * DINNER);

  float An = -__expf(A_log[d * DSTATE + n]);    // A = -exp(A_log)
  float Dd = Dp[d];
  const float* dtrow = dtf  + ((long)b * DINNER + d) * LTOK;
  const float* xrow  = xc   + ((long)b * DINNER + d) * LTOK;
  const float* zrow  = xz   + ((long)b * 2 * DINNER + DINNER + d) * LTOK;
  const float* Brow  = xdbl + ((long)b * 38 + DTRANK + n) * LTOK;
  const float* Crow  = xdbl + ((long)b * 38 + DTRANK + DSTATE + n) * LTOK;
  float* yrow = ysum + ((long)b * DINNER + d) * LTOK;

  float h = 0.f;
  for (int t = 0; t < LTOK; ++t) {
    int s = src_of(t, dir);
    if (t + 8 < LTOK) {
      int sp = src_of(t + 8, dir);
      __builtin_prefetch(&dtrow[sp], 0, 0);     // global_prefetch_b8
      __builtin_prefetch(&Brow[sp], 0, 0);
    }
    float dt = dtrow[s];
    float xv = xrow[s];
    float dA = __expf(dt * An);
    h = fmaf(dA, h, dt * Brow[s] * xv);         // h = dA*h + dt*B*x
    float y = h * Crow[s];
    y += __shfl_xor(y, 1, 32);
    y += __shfl_xor(y, 2, 32);
    y += __shfl_xor(y, 4, 32);
    y += __shfl_xor(y, 8, 32);
    if (n == 0) {
      float z = zrow[s];
      float yy = y + xv * Dd;
      yy *= z / (1.0f + __expf(-z));            // * silu(z)
      yrow[s] += yy;
    }
  }
}

__global__ void k_zero(float* __restrict__ p, long n) {
  long i = (long)blockIdx.x * blockDim.x + threadIdx.x;
  if (i < n) p[i] = 0.f;
}
__global__ void k_add(const float* __restrict__ a, const float* __restrict__ b,
                      float* __restrict__ o, long n) {
  long i = (long)blockIdx.x * blockDim.x + threadIdx.x;
  if (i < n) o[i] = a[i] + b[i];
}

// ---------------------------------------------------------------------------
// Host orchestration
// ---------------------------------------------------------------------------
static inline void gemm(hipStream_t st,
                        const float* A, long sAm, long sAk, long sAb,
                        const float* B, long sBk, long sBn, long sBb,
                        float* C, long sCm, long sCn, long sCb,
                        const float* bias, const float* res, long sRb,
                        int M, int N, int K, int ep)
{
  dim3 grid((N + 31) / 32, (M + 31) / 32, BSZ);
  k_gemm_wmma<<<grid, 128, 0, st>>>(A, sAm, sAk, sAb, B, sBk, sBn, sBb,
                                    C, sCm, sCn, sCb, bias, res, sRb,
                                    M, N, K, ep);
}

extern "C" void kernel_launch(void* const* d_in, const int* in_sizes, int n_in,
                              void* d_out, int out_size, void* d_ws, size_t ws_size,
                              hipStream_t stream)
{
  (void)in_sizes; (void)n_in; (void)out_size; (void)ws_size;
  const float* const* F = (const float* const*)d_in;

  // ---- input unpack (setup_inputs dict order, depth-first) ----
  const float* x_in  = F[0];
  const float* w1a = F[1],  *b1a = F[2],  *g1aw = F[3],  *g1ab = F[4];
  const float* w1b = F[5],  *b1b = F[6],  *g1bw = F[7],  *g1bb = F[8];
  const float* w2  = F[9],  *b2  = F[10], *g2w  = F[11], *g2b  = F[12];
  const float* w3  = F[13], *b3  = F[14], *g3w  = F[15], *g3b  = F[16];
  const float* ln_w = F[17], *ln_b = F[18];
  const float* in_proj  = F[19];
  const float* out_proj = F[20];
  const int ssm_base[3] = {21, 28, 35};   // conv_w,conv_b,xproj_w,dtproj_w,dtproj_b,A_log,D
  const float* gnw = F[42], *gnb = F[43];
  const float* mw1 = F[44], *mb1 = F[45], *mw2 = F[46], *mb2 = F[47];
  float* out = (float*)d_out;

  // ---- workspace layout (floats); ~12.13M floats ~= 48.5 MB ----
  float* ws   = (float*)d_ws;
  float* xg   = ws;                    // (B,96,L)  GSC out / post-mamba residual
  float* t0   = xg   + 786432;         // (B,L,96) xn | MLP-GN out (B,96,L)
  float* xz   = t0   + 786432;         // (B,384,L)
  float* ysum = xz   + 3145728;        // (B,192,L)
  float* xc   = ysum + 1572864;        // (B,192,L)
  float* xdbl = xc   + 1572864;        // (B,38,L)
  float* dtf  = xdbl + 327680;         // (B,192,L)
  float* tA   = dtf  + 1572864;        // scratch (up to B,192,L)
  float* tB   = tA   + 1572864;        // scratch (B,96,L)

  const long L = LTOK;
  const int mid = CDIM / 4;            // 24

  // ================= GSC =================
  k_conv3_wmma<<<dim3(LTOK / 32, (mid + 31) / 32, BSZ), 128, 0, stream>>>(
      x_in, w1a, b1a, tA, CDIM, mid);
  k_groupnorm<<<dim3(BSZ * 8), 256, 0, stream>>>(tA, tA, g1aw, g1ab, nullptr, mid, 8, 1);
  k_conv3_wmma<<<dim3(LTOK / 32, (CDIM + 31) / 32, BSZ), 128, 0, stream>>>(
      tA, w1b, b1b, tB, mid, CDIM);
  k_groupnorm<<<dim3(BSZ * 8), 256, 0, stream>>>(tB, tB, g1bw, g1bb, nullptr, CDIM, 8, 1);
  // h2 = relu(GN(1x1 conv2(x)))
  gemm(stream, w2, CDIM, 1, 0, x_in, L, 1, (long)CDIM * L,
       tA, L, 1, (long)CDIM * L, b2, nullptr, 0, CDIM, LTOK, CDIM, EP_BIAS);
  k_groupnorm<<<dim3(BSZ * 8), 256, 0, stream>>>(tA, tA, g2w, g2b, nullptr, CDIM, 8, 1);
  // s = h + h2
  k_add<<<dim3((786432 + 255) / 256), 256, 0, stream>>>(tB, tA, tB, 786432);
  // xg = relu(GN(1x1 conv3(s))) + x
  gemm(stream, w3, CDIM, 1, 0, tB, L, 1, (long)CDIM * L,
       tA, L, 1, (long)CDIM * L, b3, nullptr, 0, CDIM, LTOK, CDIM, EP_BIAS);
  k_groupnorm<<<dim3(BSZ * 8), 256, 0, stream>>>(tA, xg, g3w, g3b, x_in, CDIM, 8, 1);

  // ================= LayerNorm + in_proj =================
  k_layernorm<<<dim3((BSZ * LTOK + 255) / 256), 256, 0, stream>>>(xg, ln_w, ln_b, t0);
  gemm(stream, in_proj, CDIM, 1, 0, t0, 1, CDIM, (long)LTOK * CDIM,
       xz, L, 1, (long)2 * DINNER * L, nullptr, nullptr, 0,
       2 * DINNER, LTOK, CDIM, EP_NONE);

  // ================= tri-directional SSM =================
  k_zero<<<dim3((1572864 + 255) / 256), 256, 0, stream>>>(ysum, 1572864);
  for (int dir = 0; dir < 3; ++dir) {
    const float* cw   = F[ssm_base[dir] + 0];
    const float* cb   = F[ssm_base[dir] + 1];
    const float* xpw  = F[ssm_base[dir] + 2];
    const float* dtw  = F[ssm_base[dir] + 3];
    const float* dtb  = F[ssm_base[dir] + 4];
    const float* Alog = F[ssm_base[dir] + 5];
    const float* Dp   = F[ssm_base[dir] + 6];

    k_dwconv_silu<<<dim3((BSZ * DINNER * LTOK + 255) / 256), 256, 0, stream>>>(
        xz, cw, cb, xc, dir);
    gemm(stream, xpw, DINNER, 1, 0, xc, L, 1, (long)DINNER * L,
         xdbl, L, 1, 38L * L, nullptr, nullptr, 0, 38, LTOK, DINNER, EP_NONE);
    gemm(stream, dtw, DTRANK, 1, 0, xdbl, L, 1, 38L * L,
         dtf, L, 1, (long)DINNER * L, dtb, nullptr, 0,
         DINNER, LTOK, DTRANK, EP_BIAS_SOFTPLUS);
    k_scan<<<dim3((BSZ * DINNER * DSTATE) / 256), 256, 0, stream>>>(
        xc, xdbl, dtf, xz, Alog, Dp, ysum, dir);
  }

  // xg = out_proj_w (96x192) * ysum + xg
  gemm(stream, out_proj, DINNER, 1, 0, ysum, L, 1, (long)DINNER * L,
       xg, L, 1, (long)CDIM * L, nullptr, xg, (long)CDIM * L,
       CDIM, LTOK, DINNER, EP_RES);

  // ================= MLP =================
  k_groupnorm<<<dim3(BSZ * 8), 256, 0, stream>>>(xg, t0, gnw, gnb, nullptr, CDIM, 8, 0);
  gemm(stream, mw1, CDIM, 1, 0, t0, L, 1, (long)CDIM * L,
       tA, L, 1, (long)DINNER * L, mb1, nullptr, 0,
       DINNER, LTOK, CDIM, EP_BIAS_GELU);
  gemm(stream, mw2, DINNER, 1, 0, tA, L, 1, (long)DINNER * L,
       out, L, 1, (long)CDIM * L, mb2, xg, (long)CDIM * L,
       CDIM, LTOK, DINNER, EP_BIAS_RES);
}
